// TransformerBlock_52140902973548
// MI455X (gfx1250) — matrix operand order repaired and checked by reading, not compiled, not run on hardware
//
#include <hip/hip_runtime.h>

// ---------------------------------------------------------------------------
// CDNA5 (gfx1250) transformer block: bf16 WMMA GEMMs + flash attention.
// Wave32; WMMA shape 16x16x32 bf16 -> f32.
// ---------------------------------------------------------------------------

typedef __attribute__((ext_vector_type(16))) __bf16   v16bf;
typedef __attribute__((ext_vector_type(8)))  float    v8f;
typedef __attribute__((ext_vector_type(4)))  unsigned v4u;

union FragBF {           // one WMMA 16-lane-striped operand (32 bytes/lane)
    v16bf bf;
    v4u   q[2];
};

union Pack8 {            // 8 bf16 <-> v4u
    v4u            q;
    unsigned short u[8];
};

__device__ __forceinline__ unsigned short f32_to_bf16(float f) {
    unsigned u = __float_as_uint(f);
    u += 0x7FFFu + ((u >> 16) & 1u);     // round-to-nearest-even
    return (unsigned short)(u >> 16);
}

__device__ __forceinline__ v8f wmma_bf16(const FragBF& a, const FragBF& b, v8f c) {
    return __builtin_amdgcn_wmma_f32_16x16x32_bf16(
        /*neg_a=*/false, a.bf, /*neg_b=*/false, b.bf,
        /*c_mod=*/(short)0, c, /*reuse_a=*/false, /*reuse_b=*/false);
}

// ---------------------------------------------------------------------------
// VALU cross-lane xor within 16-lane half-groups (no LDS: DPP / permlane16).
// Keeps the LDS pipe free for WMMA fragment traffic in the attention loop.
// ---------------------------------------------------------------------------
__device__ __forceinline__ float lxor1(float x) {   // quad_perm [1,0,3,2]
    return __int_as_float(
        __builtin_amdgcn_mov_dpp(__float_as_int(x), 0xB1, 0xF, 0xF, true));
}
__device__ __forceinline__ float lxor2(float x) {   // quad_perm [2,3,0,1]
    return __int_as_float(
        __builtin_amdgcn_mov_dpp(__float_as_int(x), 0x4E, 0xF, 0xF, true));
}
__device__ __forceinline__ float lxor4(float x) {   // dpp8 [4,5,6,7,0,1,2,3]
    return __int_as_float(
        __builtin_amdgcn_mov_dpp8(__float_as_int(x), 0x688FAC));
}
__device__ __forceinline__ float lxor8(float x) {   // permlane16, lane ^= 8
    int i = __float_as_int(x);
    return __int_as_float(
        __builtin_amdgcn_permlane16(i, i, 0xFEDCBA98u, 0x76543210u, false, false));
}
__device__ __forceinline__ float red16_max(float v) {
    v = fmaxf(v, lxor1(v)); v = fmaxf(v, lxor2(v));
    v = fmaxf(v, lxor4(v)); v = fmaxf(v, lxor8(v));
    return v;
}
__device__ __forceinline__ float red16_sum(float v) {
    v += lxor1(v); v += lxor2(v); v += lxor4(v); v += lxor8(v);
    return v;
}

// ---------------------------------------------------------------------------
// fp32 -> bf16 conversion (grid-stride)
// ---------------------------------------------------------------------------
__global__ void cvt_bf16_kernel(const float* __restrict__ in,
                                unsigned short* __restrict__ out, int n) {
    for (int i = blockIdx.x * blockDim.x + threadIdx.x; i < n;
         i += gridDim.x * blockDim.x)
        out[i] = f32_to_bf16(in[i]);
}

// ---------------------------------------------------------------------------
// Generic bf16 GEMM: C[M,N] = A[M,K] @ B[K,N] (+bias, +relu), f32 or bf16 out.
// Block tile 128x256, 8 waves (2x4), wave tile 64x64 = 4x4 WMMA tiles.
// FLAGS template param -> straight-line, branch-free epilogue (no spills).
// Requires M%128==0, N%256==0, K%32==0 (true for all calls here).
// FLAGS: 1=relu, 2=store bf16, 4=add bias
// ---------------------------------------------------------------------------
template <int FLAGS>
__global__ __launch_bounds__(256) void gemm_bf16_kernel(
    const unsigned short* __restrict__ A, const unsigned short* __restrict__ B,
    const float* __restrict__ bias, void* __restrict__ C,
    int M, int N, int K)
{
    __shared__ __align__(16) unsigned short sA[128][40];    // [m][k], 16B chunks
    __shared__ __align__(16) unsigned short sB[32][264];    // [k][n], 16B aligned rows

    const int bn = blockIdx.x, bm = blockIdx.y;
    const int w = threadIdx.x >> 5, lane = threadIdx.x & 31;
    const int wm = w >> 2, wn = w & 3;          // 2 x 4 wave grid
    const int g = lane >> 4, ln = lane & 15;

    v8f acc[4][4] = {};

    for (int k0 = 0; k0 < K; k0 += 32) {
        // cooperative staging: A 128x32 (512 x 16B), B 32x256 (1024 x 16B)
        #pragma unroll
        for (int p = 0; p < 2; ++p) {
            int c = threadIdx.x + p * 256;
            int ar = c >> 2, ac = (c & 3) * 8;
            *(v4u*)&sA[ar][ac] =
                *(const v4u*)(A + (size_t)(bm * 128 + ar) * K + k0 + ac);
        }
        #pragma unroll
        for (int p = 0; p < 4; ++p) {
            int c = threadIdx.x + p * 256;
            int br = c >> 5, bc = (c & 31) * 8;
            *(v4u*)&sB[br][bc] =
                *(const v4u*)(B + (size_t)(k0 + br) * N + bn * 256 + bc);
        }
        __syncthreads();

        FragBF fa[4], fb[4];
        #pragma unroll
        for (int mt = 0; mt < 4; ++mt) {
            int row = wm * 64 + mt * 16 + ln;
            fa[mt].q[0] = *(const v4u*)&sA[row][g * 8];        // K = g*8 .. +7
            fa[mt].q[1] = *(const v4u*)&sA[row][16 + g * 8];   // K = 16+g*8 .. +7
        }
        #pragma unroll
        for (int nt = 0; nt < 4; ++nt) {
            int col = wn * 64 + nt * 16;
            Pack8 t0, t1;                                      // N = col+ln, K = g*8+s / 16+g*8+s
            #pragma unroll
            for (int s = 0; s < 8; ++s) { t0.u[s] = sB[g * 8 + s][col + ln]; t1.u[s] = sB[16 + g * 8 + s][col + ln]; }
            fb[nt].q[0] = t0.q;
            fb[nt].q[1] = t1.q;
        }
        #pragma unroll
        for (int mt = 0; mt < 4; ++mt)
            #pragma unroll
            for (int nt = 0; nt < 4; ++nt)
                acc[mt][nt] = wmma_bf16(fa[mt], fb[nt], acc[mt][nt]);
        __syncthreads();
    }

    float*          Cf = (float*)C;
    unsigned short* Cb = (unsigned short*)C;
    #pragma unroll
    for (int mt = 0; mt < 4; ++mt)
        #pragma unroll
        for (int nt = 0; nt < 4; ++nt) {
            int col = bn * 256 + wn * 64 + nt * 16 + ln;
            float bv = (FLAGS & 4) ? bias[col] : 0.0f;
            #pragma unroll
            for (int r = 0; r < 8; ++r) {
                int row = bm * 128 + wm * 64 + mt * 16 + g * 8 + r;
                float v = acc[mt][nt][r] + bv;
                if (FLAGS & 1) v = fmaxf(v, 0.0f);
                if (FLAGS & 2) Cb[(size_t)row * N + col] = f32_to_bf16(v);
                else           Cf[(size_t)row * N + col] = v;
            }
        }
}

// ---------------------------------------------------------------------------
// Fused flash attention. qkv: bf16 [4096, 3072] (token-major; cols = 3*H with
// [q|k|v] each split into 16 heads x 64). Each wave: one 16-row query tile.
// Block = 8 waves covering 8 consecutive q-tiles of one (batch, head), so all
// waves share the K/V tiles staged in LDS. Online softmax, scale = 1/sqrt(H).
// ---------------------------------------------------------------------------
__global__ __launch_bounds__(256) void attn_kernel(
    const unsigned short* __restrict__ qkv, float* __restrict__ out)
{
    const int bh   = blockIdx.x >> 4;       // 0..31
    const int b    = bh >> 4;               // 0..1
    const int n    = bh & 15;               // head
    const int qblk = blockIdx.x & 15;
    const int w    = threadIdx.x >> 5, lane = threadIdx.x & 31;
    const int qt   = qblk * 8 + w;          // q-tile 0..127
    const int g    = lane >> 4, ln = lane & 15;
    const float scale = 0.03125f;           // 1/sqrt(1024) (faithful quirk)

    __shared__ __align__(16) unsigned short sK[64][40];     // [d][s_local] (transposed)
    __shared__ __align__(16) unsigned short sV[32][72];     // [s_local][d]
    __shared__ __align__(16) unsigned short sP[8][16][40];  // per-wave P relayout

    // Q fragments: A operand, 16 rows x 64 (two 32-K steps)
    FragBF aq[2];
    {
        const unsigned short* qp =
            qkv + (size_t)(b * 2048 + qt * 16 + ln) * 3072 + n * 64;
        #pragma unroll
        for (int kq = 0; kq < 2; ++kq) {
            aq[kq].q[0] = *(const v4u*)(qp + kq * 32 + g * 8);
            aq[kq].q[1] = *(const v4u*)(qp + kq * 32 + 16 + g * 8);
        }
    }

    float m8[8], l8[8];
    v8f   O[4] = {};
    #pragma unroll
    for (int i = 0; i < 8; ++i) { m8[i] = -1e30f; l8[i] = 0.0f; }

    for (int kt = 0; kt < 64; ++kt) {       // 2048 keys / 32 per step
        const int s0 = kt * 32;
        // stage K (transposed) and V tiles: 32 rows x 64 cols each
        {
            int sr = threadIdx.x >> 3, dc = (threadIdx.x & 7) * 8;
            const unsigned short* base =
                qkv + (size_t)(b * 2048 + s0 + sr) * 3072 + n * 64 + dc;
            Pack8 kp; kp.q = *(const v4u*)(base + 1024);
            #pragma unroll
            for (int j = 0; j < 8; ++j) sK[dc + j][sr] = kp.u[j];
            *(v4u*)&sV[sr][dc] = *(const v4u*)(base + 2048);
        }
        __syncthreads();

        // scores: two 16x16 D tiles covering 32 key columns
        v8f sc[2];
        #pragma unroll
        for (int st = 0; st < 2; ++st) {
            v8f c = {};
            FragBF bk;
            #pragma unroll
            for (int j = 0; j < 2; ++j) {   // contraction over d = 64 (2 x 32)
                Pack8 t0, t1;
                #pragma unroll
                for (int s = 0; s < 8; ++s) { t0.u[s] = sK[j * 32 + g * 8 + s][st * 16 + ln]; t1.u[s] = sK[j * 32 + 16 + g * 8 + s][st * 16 + ln]; }
                bk.q[0] = t0.q;
                bk.q[1] = t1.q;
                c = wmma_bf16(aq[j], bk, c);
            }
            sc[st] = c * scale;
        }

        // online softmax statistics; reductions across 16-lane half-groups
        // use VALU DPP/permlane (no LDS contention with fragment loads).
        float al[8];
        #pragma unroll
        for (int i = 0; i < 8; ++i) {
            float mx = red16_max(fmaxf(sc[0][i], sc[1][i]));
            float mn = fmaxf(m8[i], mx);
            al[i] = __expf(m8[i] - mn);
            m8[i] = mn;
            sc[0][i] = __expf(sc[0][i] - mn);
            sc[1][i] = __expf(sc[1][i] - mn);
            float sm = red16_sum(sc[0][i] + sc[1][i]);
            l8[i] = l8[i] * al[i] + sm;
        }
        #pragma unroll
        for (int dt = 0; dt < 4; ++dt)
            #pragma unroll
            for (int i = 0; i < 8; ++i) O[dt][i] *= al[i];

        // P: D layout -> A layout via per-wave LDS tile (wave-synchronous)
        #pragma unroll
        for (int st = 0; st < 2; ++st)
            #pragma unroll
            for (int i = 0; i < 8; ++i)
                sP[w][i + g * 8][st * 16 + ln] = f32_to_bf16(sc[st][i]);
        asm volatile("s_wait_dscnt 0" ::: "memory");
        FragBF ap;
        ap.q[0] = *(const v4u*)&sP[w][ln][g * 8];
        ap.q[1] = *(const v4u*)&sP[w][ln][16 + g * 8];

        // O += P @ V   (4 d-tiles of 16)
        #pragma unroll
        for (int dt = 0; dt < 4; ++dt) {
            FragBF bv;
            Pack8 t0, t1;
            #pragma unroll
            for (int s = 0; s < 8; ++s) { t0.u[s] = sV[g * 8 + s][dt * 16 + ln]; t1.u[s] = sV[16 + g * 8 + s][dt * 16 + ln]; }
            bv.q[0] = t0.q;
            bv.q[1] = t1.q;
            O[dt] = wmma_bf16(ap, bv, O[dt]);
        }
        __syncthreads();
    }

    #pragma unroll
    for (int i = 0; i < 8; ++i) {
        float inv = 1.0f / l8[i];
        size_t row = (size_t)(b * 2048 + qt * 16 + i + g * 8) * 1024;
        #pragma unroll
        for (int dt = 0; dt < 4; ++dt)
            out[row + n * 64 + dt * 16 + ln] = O[dt][i] * inv;
    }
}

// ---------------------------------------------------------------------------
// out[tok] = resid[tok] + LayerNorm(a[tok]) * gamma + beta   (H = 1024)
// one block per token, 256 threads x 4 elements
// ---------------------------------------------------------------------------
__global__ __launch_bounds__(256) void ln_residual_kernel(
    const float* __restrict__ resid, const float* __restrict__ a,
    const float* __restrict__ gamma, const float* __restrict__ beta,
    float* __restrict__ out)
{
    const size_t base = (size_t)blockIdx.x * 1024;
    float v[4], s = 0.0f, ss = 0.0f;
    #pragma unroll
    for (int j = 0; j < 4; ++j) {
        v[j] = a[base + threadIdx.x + j * 256];
        s += v[j]; ss += v[j] * v[j];
    }
    #pragma unroll
    for (int m = 1; m < 32; m <<= 1) {
        s  += __shfl_xor(s, m, 32);
        ss += __shfl_xor(ss, m, 32);
    }
    __shared__ float rs[8], rss[8];
    const int w = threadIdx.x >> 5, lane = threadIdx.x & 31;
    if (lane == 0) { rs[w] = s; rss[w] = ss; }
    __syncthreads();
    s = 0.0f; ss = 0.0f;
    #pragma unroll
    for (int i = 0; i < 8; ++i) { s += rs[i]; ss += rss[i]; }
    const float mu  = s * (1.0f / 1024.0f);
    const float var = ss * (1.0f / 1024.0f) - mu * mu;
    const float inv = rsqrtf(var + 1e-5f);
    #pragma unroll
    for (int j = 0; j < 4; ++j) {
        int idx = threadIdx.x + j * 256;
        out[base + idx] = resid[base + idx] + (v[j] - mu) * inv * gamma[idx] + beta[idx];
    }
}

// ---------------------------------------------------------------------------
extern "C" void kernel_launch(void* const* d_in, const int* in_sizes, int n_in,
                              void* d_out, int out_size, void* d_ws, size_t ws_size,
                              hipStream_t stream)
{
    const float* x     = (const float*)d_in[0];
    const float* w_qkv = (const float*)d_in[1];
    const float* ln1_g = (const float*)d_in[2];
    const float* ln1_b = (const float*)d_in[3];
    const float* w1    = (const float*)d_in[4];
    const float* b1    = (const float*)d_in[5];
    const float* w2    = (const float*)d_in[6];
    const float* b2    = (const float*)d_in[7];
    const float* ln2_g = (const float*)d_in[8];
    const float* ln2_b = (const float*)d_in[9];
    float* out = (float*)d_out;

    const int M = 4096, H = 1024, QKV = 3072, DF = 4096;

    char* p = (char*)d_ws;
    unsigned short* xb    = (unsigned short*)p; p += (size_t)M * H   * 2;
    unsigned short* wqkvb = (unsigned short*)p; p += (size_t)H * QKV * 2;
    unsigned short* qkvb  = (unsigned short*)p; p += (size_t)M * QKV * 2;
    float*          attn  = (float*)p;          p += (size_t)M * H   * 4;
    float*          x1    = (float*)p;          p += (size_t)M * H   * 4;
    unsigned short* x1b   = (unsigned short*)p; p += (size_t)M * H   * 2;
    unsigned short* w1b   = (unsigned short*)p; p += (size_t)H * DF  * 2;
    unsigned short* h1b   = (unsigned short*)p; p += (size_t)M * DF  * 2;
    unsigned short* w2b   = (unsigned short*)p; p += (size_t)DF * H  * 2;
    float*          ff    = (float*)p;          p += (size_t)M * H   * 4;

    // fp32 -> bf16 conversions
    cvt_bf16_kernel<<<2048, 256, 0, stream>>>(x,     xb,    M * H);
    cvt_bf16_kernel<<<2048, 256, 0, stream>>>(w_qkv, wqkvb, H * QKV);
    cvt_bf16_kernel<<<2048, 256, 0, stream>>>(w1,    w1b,   H * DF);
    cvt_bf16_kernel<<<2048, 256, 0, stream>>>(w2,    w2b,   DF * H);

    // qkv = x @ w_qkv   (store bf16 for attention)
    gemm_bf16_kernel<2><<<dim3(QKV / 256, M / 128), 256, 0, stream>>>(
        xb, wqkvb, nullptr, qkvb, M, QKV, H);

    // fused flash attention -> attn (f32 [4096,1024])
    attn_kernel<<<512, 256, 0, stream>>>(qkvb, attn);

    // x1 = x + LN(attn)
    ln_residual_kernel<<<M, 256, 0, stream>>>(x, attn, ln1_g, ln1_b, x1);
    cvt_bf16_kernel<<<2048, 256, 0, stream>>>(x1, x1b, M * H);

    // h1 = relu(x1 @ w1 + b1)   (store bf16)
    gemm_bf16_kernel<1 | 2 | 4><<<dim3(DF / 256, M / 128), 256, 0, stream>>>(
        x1b, w1b, b1, h1b, M, DF, H);

    // ff = h1 @ w2 + b2   (f32)
    gemm_bf16_kernel<4><<<dim3(H / 256, M / 128), 256, 0, stream>>>(
        h1b, w2b, b2, ff, M, H, DF);

    // out = x1 + LN(ff)
    ln_residual_kernel<<<M, 256, 0, stream>>>(x1, ff, ln2_g, ln2_b, out);
}
